// SparseSDFVQVAE_75539884802813
// MI455X (gfx1250) — matrix operand order repaired and checked by reading, not compiled, not run on hardware
//
#include <hip/hip_runtime.h>
#include <hip/hip_bf16.h>

typedef _Float16 v16h __attribute__((ext_vector_type(16)));
typedef _Float16 v8h  __attribute__((ext_vector_type(8)));
typedef float    v8f  __attribute__((ext_vector_type(8)));
typedef float    v4f  __attribute__((ext_vector_type(4)));

#define DD 64

// ---------------------------------------------------------------------------
// Prep: codebook f32 -> f16 copy (WMMA B operand) + exact f32 norms ||e||^2
// ---------------------------------------------------------------------------
__global__ void vq_prep_kernel(const float* __restrict__ cb,
                               _Float16* __restrict__ cb16,
                               float* __restrict__ enorm, int K) {
  int k = blockIdx.x * blockDim.x + threadIdx.x;
  if (k >= K) return;
  const float* row = cb + (size_t)k * DD;
  _Float16* out = cb16 + (size_t)k * DD;
  float s = 0.f;
#pragma unroll
  for (int j = 0; j < DD; j += 4) {
    v4f v = *(const v4f*)(row + j);
    s += v.x * v.x + v.y * v.y + v.z * v.z + v.w * v.w;
    out[j + 0] = (_Float16)v.x;
    out[j + 1] = (_Float16)v.y;
    out[j + 2] = (_Float16)v.z;
    out[j + 3] = (_Float16)v.w;
  }
  enorm[k] = s;
}

// Zero the two loss accumulators (harness poisons d_out with 0xAA).
__global__ void vq_zero_kernel(float* p) { p[0] = 0.f; p[1] = 0.f; }

// Async global->LDS copy (CDNA5, ASYNCcnt-tracked). LDS address = low 32 bits
// of the generic pointer (flat LDS aperture truncates to wave-relative offset).
__device__ __forceinline__ void async_copy_b128(void* lds, const void* gmem) {
  unsigned l = (unsigned)(unsigned long long)(uintptr_t)lds;
  unsigned long long g = (unsigned long long)(uintptr_t)gmem;
  asm volatile("global_load_async_to_lds_b128 %0, %1, off"
               :: "v"(l), "v"(g) : "memory");
}
__device__ __forceinline__ void async_copy_b32(void* lds, const void* gmem) {
  unsigned l = (unsigned)(unsigned long long)(uintptr_t)lds;
  unsigned long long g = (unsigned long long)(uintptr_t)gmem;
  asm volatile("global_load_async_to_lds_b32 %0, %1, off"
               :: "v"(l), "v"(g) : "memory");
}
__device__ __forceinline__ void wait_async0() {
  asm volatile("s_wait_asynccnt 0x0" ::: "memory");
}

// ---------------------------------------------------------------------------
// Main: fused (z @ E^T via WMMA) + argmin + gather + MSE.
// Block = 256 threads (8 waves); wave w owns rows [blockIdx*128 + 16w, +16).
// K scanned in 32-code stages, double-buffered in LDS via async copies:
//   issue(stage+1) -> compute(stage): 4x v_wmma -> s_wait_asynccnt 0 -> barrier
// ---------------------------------------------------------------------------
__launch_bounds__(256)
__global__ void vq_main_kernel(const float* __restrict__ z,
                               const float* __restrict__ cb,
                               const _Float16* __restrict__ cb16,
                               const float* __restrict__ enorm,
                               float* __restrict__ out, int Nn, int K) {
  __shared__ __attribute__((aligned(16))) _Float16 sB[2][32][72];  // 32 codes x 64 halfs (+8 pad)
  __shared__ float sN[2][32];

  const int t    = threadIdx.x;
  const int wave = t >> 5;
  const int lane = t & 31;
  const int kg   = lane >> 4;   // K-half group (0/1) per WMMA operand layout
  const int ln   = lane & 15;
  const int rowBase = blockIdx.x * 128 + wave * 16;

  // --- A operands (D=64 -> two K=32 chunks), f32 -> f16, exact 16-bit
  //     A-matrix VGPR layout: halves j<8 -> K=j+8*kg, j>=8 -> K=j+8+8*kg.
  const float* zr = z + (size_t)(rowBase + ln) * DD;
  v16h aOp[2];
#pragma unroll
  for (int c = 0; c < 2; ++c) {
    int s1 = 32 * c + 8 * kg;
    int s2 = 32 * c + 16 + 8 * kg;
    v4f r0 = *(const v4f*)(zr + s1);
    v4f r1 = *(const v4f*)(zr + s1 + 4);
    v4f r2 = *(const v4f*)(zr + s2);
    v4f r3 = *(const v4f*)(zr + s2 + 4);
    v16h a;
    a[0]  = (_Float16)r0.x; a[1]  = (_Float16)r0.y; a[2]  = (_Float16)r0.z; a[3]  = (_Float16)r0.w;
    a[4]  = (_Float16)r1.x; a[5]  = (_Float16)r1.y; a[6]  = (_Float16)r1.z; a[7]  = (_Float16)r1.w;
    a[8]  = (_Float16)r2.x; a[9]  = (_Float16)r2.y; a[10] = (_Float16)r2.z; a[11] = (_Float16)r2.w;
    a[12] = (_Float16)r3.x; a[13] = (_Float16)r3.y; a[14] = (_Float16)r3.z; a[15] = (_Float16)r3.w;
    aOp[c] = a;
  }

  float bestVal[8];
  int   bestIdx[8];
#pragma unroll
  for (int v = 0; v < 8; ++v) { bestVal[v] = 3.4e38f; bestIdx[v] = 0; }

  const int nStages = K >> 5;          // 32 codes per stage
  const int ldCode  = t >> 3;          // 0..31
  const int ldH8    = (t & 7) << 3;    // 0,8,...,56 halfs (16B chunks)

  // Prologue: stage 0 into buffer 0.
  async_copy_b128(&sB[0][ldCode][ldH8],
                  cb16 + ((size_t)ldCode * DD + ldH8));
  if (t < 32) async_copy_b32(&sN[0][t], enorm + t);
  wait_async0();
  __syncthreads();

  for (int st = 0; st < nStages; ++st) {
    const int buf = st & 1;
    if (st + 1 < nStages) {   // overlap next stage's copy with this stage's WMMAs
      async_copy_b128(&sB[buf ^ 1][ldCode][ldH8],
                      cb16 + ((size_t)((st + 1) * 32 + ldCode) * DD + ldH8));
      if (t < 32) async_copy_b32(&sN[buf ^ 1][t], enorm + (st + 1) * 32 + t);
    }

#pragma unroll
    for (int sub = 0; sub < 2; ++sub) {  // two 16-code WMMA tiles per stage
      const int code0 = sub * 16;
      // B operands from LDS: lane = code (ln), features 32c + 8*kg + j (j<8)
      // and 32c + 16 + 8*kg + (j-8) (j>=8), matching the A layout
      int boff = 8 * kg;
      v8h b0l = *(const v8h*)(&sB[buf][code0 + ln][boff]);
      v8h b0h = *(const v8h*)(&sB[buf][code0 + ln][boff + 16]);
      v8h b1l = *(const v8h*)(&sB[buf][code0 + ln][32 + boff]);
      v8h b1h = *(const v8h*)(&sB[buf][code0 + ln][32 + boff + 16]);
      v16h b0 = __builtin_shufflevector(b0l, b0h, 0,1,2,3,4,5,6,7,8,9,10,11,12,13,14,15);
      v16h b1 = __builtin_shufflevector(b1l, b1h, 0,1,2,3,4,5,6,7,8,9,10,11,12,13,14,15);

      v8f acc = {0.f, 0.f, 0.f, 0.f, 0.f, 0.f, 0.f, 0.f};
      acc = __builtin_amdgcn_wmma_f32_16x16x32_f16(false, aOp[0], false, b0,
                                                   (short)0, acc, false, false);
      acc = __builtin_amdgcn_wmma_f32_16x16x32_f16(false, aOp[1], false, b1,
                                                   (short)0, acc, false, false);

      // cost[m][n] = ||e_n||^2 - 2*dot  (||z_m||^2 constant per row -> dropped)
      float nrm = sN[buf][code0 + ln];
      int   idx = st * 32 + code0 + ln;
#pragma unroll
      for (int v = 0; v < 8; ++v) {
        float cost = fmaf(-2.f, acc[v], nrm);
        if (cost < bestVal[v]) { bestVal[v] = cost; bestIdx[v] = idx; }  // strict <: first-min
      }
    }

    if (st + 1 < nStages) wait_async0();  // next buffer landed (this wave's pieces)
    __syncthreads();                      // ...and everyone else's; reads of buf done
  }

  // Min+argmin butterfly across the 16-lane N group (xor masks stay in-group).
#pragma unroll
  for (int v = 0; v < 8; ++v) {
#pragma unroll
    for (int off = 1; off < 16; off <<= 1) {
      float ov = __shfl_xor(bestVal[v], off, 32);
      int   oi = __shfl_xor(bestIdx[v], off, 32);
      if (ov < bestVal[v] || (ov == bestVal[v] && oi < bestIdx[v])) {
        bestVal[v] = ov; bestIdx[v] = oi;
      }
    }
  }

  // Epilogue: gather quantized rows (f32, exact), indices, fused MSE.
  const size_t ND = (size_t)Nn * DD;
  float lsum = 0.f;
#pragma unroll
  for (int v = 0; v < 8; ++v) {
    int row = rowBase + v + 8 * kg;      // slot v -> row v (lanes 0-15) / v+8 (16-31)
    int col = ln * 4;                    // 16 lanes x float4 = 64 cols
    v4f qv = *(const v4f*)(cb + (size_t)bestIdx[v] * DD + col);
    v4f zv = *(const v4f*)(z + (size_t)row * DD + col);
    float dx = qv.x - zv.x, dy = qv.y - zv.y, dz = qv.z - zv.z, dw = qv.w - zv.w;
    lsum += dx * dx + dy * dy + dz * dz + dw * dw;
    *(v4f*)(out + (size_t)row * DD + col) = qv;     // quantized = z + sg(q-z) = q
    if (ln == 0) out[ND + 2 + row] = (float)bestIdx[v];
  }
#pragma unroll
  for (int off = 1; off < 32; off <<= 1) lsum += __shfl_xor(lsum, off, 32);
  if (lane == 0) {
    float m = lsum / (float)ND;          // vq_loss == commitment_loss in forward
    atomicAdd(out + ND, m);
    atomicAdd(out + ND + 1, m);
  }
}

// ---------------------------------------------------------------------------
extern "C" void kernel_launch(void* const* d_in, const int* in_sizes, int n_in,
                              void* d_out, int out_size, void* d_ws, size_t ws_size,
                              hipStream_t stream) {
  const float* z  = (const float*)d_in[0];
  const float* cb = (const float*)d_in[1];
  int Nn = in_sizes[0] / DD;   // 32768
  int K  = in_sizes[1] / DD;   // 8192

  _Float16* cb16 = (_Float16*)d_ws;                                   // K*64*2 = 1 MB
  float* enorm = (float*)((char*)d_ws + (size_t)K * DD * sizeof(_Float16));
  float* out = (float*)d_out;

  vq_prep_kernel<<<(K + 255) / 256, 256, 0, stream>>>(cb, cb16, enorm, K);
  vq_zero_kernel<<<1, 1, 0, stream>>>(out + (size_t)Nn * DD);
  vq_main_kernel<<<Nn / 128, 256, 0, stream>>>(z, cb, cb16, enorm, out, Nn, K);
}
